// WindowAttention_68255620268907
// MI455X (gfx1250) — matrix operand order repaired and checked by reading, not compiled, not run on hardware
//
#include <hip/hip_runtime.h>
#include <hip/hip_bf16.h>

typedef __attribute__((ext_vector_type(16))) _Float16 v16h;
typedef __attribute__((ext_vector_type(8)))  _Float16 v8h;
typedef __attribute__((ext_vector_type(8)))  float    v8f;

#define H       128
#define S_LEN   4096
#define WIN     127
#define QTILE   128     // queries per block
#define KSPAN   384     // staged key window per block
#define QSTR    136     // padded LDS stride (halfs) for Q rows
#define KSTR    136     // padded LDS stride (halfs) for K rows
#define VSTR    392     // padded LDS stride (halfs) for Vt rows (KSPAN + 8)
#define PSTR    40      // padded LDS stride (halfs) for P rows
#define SCALE   0.08838834764831845f   // 128^-0.5
#define L2_10K  13.287712379549449f    // log2(10000)

// ---- WMMA fragment loaders (layouts per CDNA5 ISA 7.12.2) ----
// A (16x32 f16): lane m=lane&15 holds row m; elems 0..7 = K{0..7}+8*hi,
// elems 8..15 = K{16..23}+8*hi.  p points at row + k0 + 8*hi.
__device__ inline v16h loadA(const _Float16* p) {
    v16h r;
    ((v8h*)&r)[0] = *(const v8h*)(p);
    ((v8h*)&r)[1] = *(const v8h*)(p + 16);
    return r;
}
// B (32x16 f16): lane n=lane&15 holds column n; elems 0..7 = K{0..7}+8*hi,
// elems 8..15 = K{16..23}+8*hi.  p points at column + k0 + 8*hi.
__device__ inline v16h loadB(const _Float16* p) {
    v16h r;
    ((v8h*)&r)[0] = *(const v8h*)(p);
    ((v8h*)&r)[1] = *(const v8h*)(p + 16);
    return r;
}

__device__ inline float rope_invfreq(int j) {
    // 10000^(-j/64) = 2^(-j/64 * log2(10000))
    return exp2f(-(float)j * (L2_10K / 64.0f));
}

__global__ __launch_bounds__(256, 1)
void swin_attn_kernel(const float* __restrict__ q,
                      const float* __restrict__ k,
                      const float* __restrict__ v,
                      float* __restrict__ out) {
    __shared__ _Float16 sQ[QTILE * QSTR];   //  34816 B
    __shared__ _Float16 sK[KSPAN * KSTR];   // 104448 B
    __shared__ _Float16 sVt[H * VSTR];      // 100352 B
    __shared__ _Float16 sP[8 * 16 * PSTR];  //  10240 B

    const int tid   = threadIdx.x;
    const int qtile = blockIdx.x;          // 0..31
    const int bh    = blockIdx.y;          // 0..63
    const int kv0   = qtile * QTILE - 128; // first staged (global) key index

    const size_t base = (size_t)bh * S_LEN * H;

    // ---------------- Stage Q (RoPE + scale -> f16) ----------------
    {
        int r  = tid >> 1;                 // query row in tile
        int j0 = (tid & 1) * 32;           // dim-pair block
        int t  = qtile * QTILE + r;
        const float* qrow = q + base + (size_t)t * H;
        #pragma unroll
        for (int jj = 0; jj < 32; jj += 4) {
            int j = j0 + jj;
            float4 x1 = *(const float4*)(qrow + j);
            float4 x2 = *(const float4*)(qrow + j + 64);
            #pragma unroll
            for (int c = 0; c < 4; ++c) {
                float a1 = (&x1.x)[c], a2 = (&x2.x)[c];
                float ang = (float)t * rope_invfreq(j + c);
                float sn, cs; __sincosf(ang, &sn, &cs);
                sQ[r * QSTR + j + c]      = (_Float16)((a1 * cs - a2 * sn) * SCALE);
                sQ[r * QSTR + j + c + 64] = (_Float16)((a2 * cs + a1 * sn) * SCALE);
            }
        }
    }
    // ---------------- Stage K (RoPE -> f16, row-major) ----------------
    #pragma unroll
    for (int it = 0; it < 3; ++it) {
        int task = tid + it * 256;         // 768 tasks
        int r  = task >> 1;
        int j0 = (task & 1) * 32;
        int u  = kv0 + r;
        if (u >= 0 && u < S_LEN) {
            const float* krow = k + base + (size_t)u * H;
            #pragma unroll
            for (int jj = 0; jj < 32; jj += 4) {
                int j = j0 + jj;
                float4 x1 = *(const float4*)(krow + j);
                float4 x2 = *(const float4*)(krow + j + 64);
                #pragma unroll
                for (int c = 0; c < 4; ++c) {
                    float a1 = (&x1.x)[c], a2 = (&x2.x)[c];
                    float ang = (float)u * rope_invfreq(j + c);
                    float sn, cs; __sincosf(ang, &sn, &cs);
                    sK[r * KSTR + j + c]      = (_Float16)(a1 * cs - a2 * sn);
                    sK[r * KSTR + j + c + 64] = (_Float16)(a2 * cs + a1 * sn);
                }
            }
        } else {
            for (int jj = 0; jj < 32; ++jj) {
                sK[r * KSTR + j0 + jj]      = (_Float16)0.f;
                sK[r * KSTR + j0 + jj + 64] = (_Float16)0.f;
            }
        }
    }
    // ---------------- Stage V transposed: sVt[d][key] ----------------
    #pragma unroll
    for (int it = 0; it < 3; ++it) {
        int task = tid + it * 256;
        int r  = task >> 1;
        int j0 = (task & 1) * 64;
        int u  = kv0 + r;
        if (u >= 0 && u < S_LEN) {
            const float* vrow = v + base + (size_t)u * H;
            #pragma unroll
            for (int jj = 0; jj < 64; jj += 4) {
                float4 x = *(const float4*)(vrow + j0 + jj);
                #pragma unroll
                for (int c = 0; c < 4; ++c)
                    sVt[(j0 + jj + c) * VSTR + r] = (_Float16)((&x.x)[c]);
            }
        } else {
            for (int jj = 0; jj < 64; ++jj)
                sVt[(j0 + jj) * VSTR + r] = (_Float16)0.f;
        }
    }
    __syncthreads();

    // ---------------- Per-wave flash-attention over the band ----------------
    const int wave = tid >> 5, lane = tid & 31;
    const int n = lane & 15, hi = lane >> 4;
    const int qr0 = wave * 16;                   // wave's local query rows
    const int t0  = qtile * QTILE + qr0;

    v8f O[8];
    float rm[8], rl[8];
    #pragma unroll
    for (int j = 0; j < 8; ++j) O[j] = {};
    #pragma unroll
    for (int i = 0; i < 8; ++i) { rm[i] = -1e30f; rl[i] = 0.f; }

    // key-groups of 32 intersecting [t0-127, t0+15+127] (local coords)
    const int g_lo = (qr0 + 1) >> 5;
    const int g_hi = (qr0 + 270) >> 5;

    _Float16* sPw = sP + wave * (16 * PSTR);
    const _Float16* qArow = sQ + (qr0 + n) * QSTR + 8 * hi;
    const _Float16* pArow = sPw + n * PSTR + 8 * hi;

    for (int g = g_lo; g <= g_hi; ++g) {
        const int lkb = g * 32;
        // ---- S tiles: S0 = keys [lkb,lkb+16), S1 = keys [lkb+16,lkb+32)
        v8f S0 = {}, S1 = {};
        const _Float16* k0row = sK + (lkb + n) * KSTR + 8 * hi;
        const _Float16* k1row = k0row + 16 * KSTR;
        #pragma unroll
        for (int kk = 0; kk < 4; ++kk) {
            v16h a  = loadA(qArow + kk * 32);
            v16h b0 = loadB(k0row + kk * 32);
            S0 = __builtin_amdgcn_wmma_f32_16x16x32_f16(false, a, false, b0,
                                                        (short)0, S0, false, false);
            v16h b1 = loadB(k1row + kk * 32);
            S1 = __builtin_amdgcn_wmma_f32_16x16x32_f16(false, a, false, b1,
                                                        (short)0, S1, false, false);
        }
        // ---- band mask + online softmax update
        const int u0 = kv0 + lkb + n;
        const int u1 = u0 + 16;
        #pragma unroll
        for (int i = 0; i < 8; ++i) {
            int t = t0 + i + 8 * hi;
            bool v0 = (u0 >= 0) && (u0 < S_LEN) && (t - u0 <= WIN) && (u0 - t <= WIN);
            bool v1 = (u1 >= 0) && (u1 < S_LEN) && (t - u1 <= WIN) && (u1 - t <= WIN);
            float s0 = v0 ? S0[i] : -1e30f;
            float s1 = v1 ? S1[i] : -1e30f;
            float mx = fmaxf(s0, s1);
            mx = fmaxf(mx, __shfl_xor(mx, 1, 32));
            mx = fmaxf(mx, __shfl_xor(mx, 2, 32));
            mx = fmaxf(mx, __shfl_xor(mx, 4, 32));
            mx = fmaxf(mx, __shfl_xor(mx, 8, 32));
            float mnew = fmaxf(rm[i], mx);
            float corr = __expf(rm[i] - mnew);
            float p0 = v0 ? __expf(s0 - mnew) : 0.f;
            float p1 = v1 ? __expf(s1 - mnew) : 0.f;
            float ps = p0 + p1;
            ps += __shfl_xor(ps, 1, 32);
            ps += __shfl_xor(ps, 2, 32);
            ps += __shfl_xor(ps, 4, 32);
            ps += __shfl_xor(ps, 8, 32);
            rl[i] = rl[i] * corr + ps;
            rm[i] = mnew;
            #pragma unroll
            for (int j = 0; j < 8; ++j) O[j][i] *= corr;
            int prow = i + 8 * hi;
            sPw[prow * PSTR + n]      = (_Float16)p0;
            sPw[prow * PSTR + n + 16] = (_Float16)p1;
        }
        // ---- O += P * V   (A = P 16x32 reused across all 8 dim-tiles)
        v16h ap = loadA(pArow);
        #pragma unroll
        for (int j = 0; j < 8; ++j) {
            v16h bv = loadB(sVt + (j * 16 + n) * VSTR + lkb + 8 * hi);
            O[j] = __builtin_amdgcn_wmma_f32_16x16x32_f16(false, ap, false, bv,
                                                          (short)0, O[j], false, false);
        }
    }

    // ---------------- normalize + store ----------------
    float* orow = out + base + (size_t)t0 * H;
    #pragma unroll
    for (int i = 0; i < 8; ++i) {
        float inv = 1.0f / rl[i];
        int r = i + 8 * hi;
        #pragma unroll
        for (int j = 0; j < 8; ++j)
            orow[r * H + j * 16 + n] = O[j][i] * inv;
    }
}

extern "C" void kernel_launch(void* const* d_in, const int* in_sizes, int n_in,
                              void* d_out, int out_size, void* d_ws, size_t ws_size,
                              hipStream_t stream) {
    const float* q = (const float*)d_in[0];
    const float* k = (const float*)d_in[1];
    const float* v = (const float*)d_in[2];
    // d_in[3] = input_mask (all ones) — unused
    float* out = (float*)d_out;
    dim3 grid(S_LEN / QTILE, 64, 1);   // 32 q-tiles x 64 batch-heads
    dim3 block(256, 1, 1);             // 8 waves
    swin_attn_kernel<<<grid, block, 0, stream>>>(q, k, v, out);
}
